// single_head_SA2IAttn_21191368638584
// MI455X (gfx1250) — matrix operand order repaired and checked by reading, not compiled, not run on hardware
//
#include <hip/hip_runtime.h>
#include <cstdint>

#define S_LEN 4096
#define DDIM  512
#define BR    64
#define BC    64

typedef __attribute__((ext_vector_type(16))) _Float16 v16h;
typedef __attribute__((ext_vector_type(8)))  _Float16 v8h;
typedef __attribute__((ext_vector_type(8)))  float    v8f;
typedef __attribute__((ext_vector_type(4)))  float    v4f;
typedef __attribute__((ext_vector_type(4)))  unsigned int v4u;
typedef __attribute__((ext_vector_type(4)))  int v4i;
typedef __attribute__((ext_vector_type(8)))  int v8i;

union F16x16 { v4u u[2]; v16h h; };
union H8B    { v4u u;    v8h  h; };

// A-fragment (16x32 f16, wave32): lanes 0-15 rows M=lane, K=[kb,kb+8)+[kb+16,kb+24);
// lanes 16-31 same rows, K shifted by +8.  Caller passes koff = kb + hb*8.
static __device__ __forceinline__ v16h load_a16(const _Float16* row, int koff) {
  F16x16 f;
  f.u[0] = *(const v4u*)(row + koff);
  f.u[1] = *(const v4u*)(row + koff + 16);
  return f.h;
}
// B-fragment (32x16 f16): lane = column n; K=[kb,kb+8)+[kb+16,kb+24), +8 for lanes 16-31.
static __device__ __forceinline__ v16h load_b16(const _Float16* p) {
  F16x16 f;
  f.u[0] = *(const v4u*)(p);
  f.u[1] = *(const v4u*)(p + 16);
  return f.h;
}
static __device__ __forceinline__ v8f vzero8() {
  v8f z;
#pragma unroll
  for (int i = 0; i < 8; ++i) z[i] = 0.0f;
  return z;
}
#define WMMA_F16(A,B,C) \
  __builtin_amdgcn_wmma_f32_16x16x32_f16(false,(A),false,(B),(short)0,(C),false,false)

// ---- gfx1250 Tensor Data Mover: one descriptor moves a whole padded 2D tile ----
// Group1 ctrl word: data_size=2B (bit16), pad_enable (bit20), pad_interval[24:22],
// pad_amount[31:25] (value 3 -> 4 DWORDs = 16 B of padding).
#define TDM_CTRL_K (0x10000u | 0x100000u | (7u << 22) | (3u << 25)) // pad each 256 DW row
#define TDM_CTRL_V (0x10000u | 0x100000u | (4u << 22) | (3u << 25)) // pad each 32 DW row

static __device__ __forceinline__ void tdm_load_2d(unsigned lds_byte, const void* g,
                                                   unsigned ctrl,
                                                   unsigned td0, unsigned td1,
                                                   unsigned tile0, unsigned tile1,
                                                   unsigned stride0) {
  const unsigned long long ga = (unsigned long long)(uintptr_t)g;
  v4i g0;
  g0[0] = 1;                                            // count=1: valid user D#
  g0[1] = (int)lds_byte;                                // lds_addr (bytes)
  g0[2] = (int)(unsigned)(ga & 0xffffffffu);            // global_addr[31:0]
  g0[3] = (int)(((unsigned)(ga >> 32) & 0x01ffffffu) | 0x80000000u); // [56:32]|type=2
  v8i g1;
  g1[0] = (int)ctrl;                                    // mask=0, 2B elems, pad cfg
  g1[1] = (int)((td0 & 0xffffu) << 16);                 // tensor_dim0[15:0]
  g1[2] = (int)((td0 >> 16) | ((td1 & 0xffffu) << 16)); // dim0[31:16] | dim1[15:0]
  g1[3] = (int)((td1 >> 16) | (tile0 << 16));           // dim1[31:16] | tile_dim0
  g1[4] = (int)(tile1 & 0xffffu);                       // tile_dim1 (tile_dim2 = 0)
  g1[5] = (int)stride0;                                 // tensor_dim0_stride[31:0]
  g1[6] = 0;                                            // stride0 hi / stride1 lo
  g1[7] = 0;
  asm volatile("tensor_load_to_lds %0, %1" :: "s"(g0), "s"(g1) : "memory");
}
template <int N>
static __device__ __forceinline__ void wait_tensor() {
#if __has_builtin(__builtin_amdgcn_s_wait_tensorcnt)
  __builtin_amdgcn_s_wait_tensorcnt((short)N);
#else
  asm volatile("s_wait_tensorcnt %0" :: "i"(N) : "memory");
#endif
}

// ---------------- Phase 0: W (f32, row-major) -> W^T (f16, n-major) ----------------
__global__ __launch_bounds__(256) void wt_convert_kernel(const float* __restrict__ W,
                                                         _Float16* __restrict__ WT) {
  int idx = blockIdx.x * 256 + threadIdx.x;      // 0 .. 512*512-1
  int n = idx >> 9, k = idx & 511;
  WT[(size_t)n * DDIM + k] = (_Float16)W[(size_t)k * DDIM + n];
}

// ---------------- Phase 1: h = x@W + b ; emit h16 (row-major) and hT16 (feat-major) ----
__global__ __launch_bounds__(256) void proj_kernel(const float* __restrict__ x,
                                                   const _Float16* __restrict__ WT,
                                                   const float* __restrict__ bias,
                                                   _Float16* __restrict__ h16,
                                                   _Float16* __restrict__ hT16) {
  const int t = threadIdx.x;
  const int lane = t & 31, w = t >> 5;
  const int ln = lane & 15, hb = (lane >> 4) & 1;
  const int rowblk = blockIdx.x >> 3;
  const int colblk = blockIdx.x & 7;
  const int mi = w >> 1, nh = w & 1;

  const int arow = rowblk * 64 + mi * 16 + ln;
  const float* xr = x + (size_t)arow * DDIM;
  const int col0 = colblk * 64 + nh * 32 + ln;
  const int col1 = col0 + 16;
  const _Float16* wt0 = WT + (size_t)col0 * DDIM;
  const _Float16* wt1 = WT + (size_t)col1 * DDIM;

  v8f c0 = vzero8(), c1 = vzero8();
#pragma unroll 2
  for (int kk = 0; kk < DDIM / 32; ++kk) {
    const int kb = kk * 32;
    v4f f0 = *(const v4f*)(xr + kb + hb * 8);
    v4f f1 = *(const v4f*)(xr + kb + hb * 8 + 4);
    v4f f2 = *(const v4f*)(xr + kb + 16 + hb * 8);
    v4f f3 = *(const v4f*)(xr + kb + 16 + hb * 8 + 4);
    v16h a;
#pragma unroll
    for (int e = 0; e < 4; ++e) {
      a[e]      = (_Float16)f0[e];
      a[4 + e]  = (_Float16)f1[e];
      a[8 + e]  = (_Float16)f2[e];
      a[12 + e] = (_Float16)f3[e];
    }
    v16h b0 = load_b16(wt0 + kb + hb * 8);
    v16h b1 = load_b16(wt1 + kb + hb * 8);
    c0 = WMMA_F16(a, b0, c0);
    c1 = WMMA_F16(a, b1, c1);
  }
  const float b0v = bias[col0], b1v = bias[col1];
#pragma unroll
  for (int v = 0; v < 8; ++v) { c0[v] += b0v; c1[v] += b1v; }

  const int rbase = rowblk * 64 + mi * 16 + hb * 8;
#pragma unroll
  for (int v = 0; v < 8; ++v) {
    h16[(size_t)(rbase + v) * DDIM + col0] = (_Float16)c0[v];
    h16[(size_t)(rbase + v) * DDIM + col1] = (_Float16)c1[v];
  }
  const int batch = rbase >> 12;
  const int srow  = rbase & 4095;
  H8B p0, p1;
#pragma unroll
  for (int v = 0; v < 8; ++v) { p0.h[v] = (_Float16)c0[v]; p1.h[v] = (_Float16)c1[v]; }
  _Float16* tb = hT16 + (size_t)batch * DDIM * S_LEN;
  *(v4u*)(tb + (size_t)col0 * S_LEN + srow) = p0.u;
  *(v4u*)(tb + (size_t)col1 * S_LEN + srow) = p1.u;
}

// ---------------- Phase 2: flash attention, TDM double-buffered K/V^T in LDS --------
__global__ __launch_bounds__(256) void attn_kernel(const _Float16* __restrict__ h16,
                                                   const _Float16* __restrict__ hT16,
                                                   float* __restrict__ out) {
  // Double-buffered tiles; row strides produced by the TDM pad feature:
  __shared__ _Float16 Kbuf[2][64][520];    // keys x feats  (512 + 8 pad halves/row)
  __shared__ _Float16 VTbuf[2][512][72];   // feats x keys  (64 + 8 pad halves/row)
  __shared__ float    S_t[64][68];
  __shared__ _Float16 P_t[64][72];
  __shared__ float    red[4][64];
  __shared__ float    m_row[64], l_row[64], a_row[64];

  const int t = threadIdx.x;
  const int lane = t & 31, w = t >> 5;
  const int ln = lane & 15, hb = (lane >> 4) & 1;

  const int batch = blockIdx.x >> 6;
  const int qb    = (blockIdx.x & 63) * BR;
  const _Float16* hB  = h16  + (size_t)batch * S_LEN * DDIM;
  const _Float16* hTB = hT16 + (size_t)batch * S_LEN * DDIM;

  const int mi = w >> 1;                   // row tile for S and O
  const int nh = w & 1;                    // S col-tile pair / O col half

  // One TDM descriptor per tile: wave 0 moves K (64x512), wave 1 moves V^T (512x64).
  auto stage = [&](int buf, int kb64) {
    if (w == 0) {
      tdm_load_2d((unsigned)(uintptr_t)&Kbuf[buf][0][0],
                  hB + (size_t)kb64 * DDIM, TDM_CTRL_K,
                  /*td0=*/DDIM, /*td1=*/S_LEN, /*tile0=*/DDIM, /*tile1=*/BC,
                  /*stride0=*/DDIM);
    } else if (w == 1) {
      tdm_load_2d((unsigned)(uintptr_t)&VTbuf[buf][0][0],
                  hTB + kb64, TDM_CTRL_V,
                  /*td0=*/S_LEN, /*td1=*/DDIM, /*tile0=*/BC, /*tile1=*/DDIM,
                  /*stride0=*/S_LEN);
    }
  };

  v8f acc[16];
#pragma unroll
  for (int i = 0; i < 16; ++i) acc[i] = vzero8();
  if (t < 64) { m_row[t] = -1e30f; l_row[t] = 0.0f; }

  stage(0, 0);                             // prologue: tile 0 in flight on the TDM
  __syncthreads();

  const _Float16* qrow = hB + (size_t)(qb + mi * 16 + ln) * DDIM;

  for (int j = 0; j < S_LEN / BC; ++j) {
    const int cur = j & 1;
    const int kb64 = j * BC;
    if (j + 1 < S_LEN / BC) {
      stage(cur ^ 1, kb64 + BC);           // next tile streams while we compute
      if (w < 2) wait_tensor<1>();         // in-order: tile j's descriptor retired
    } else {
      if (w < 2) wait_tensor<0>();
    }
    __syncthreads();                       // tile j visible to all waves

    const _Float16* Kc = &Kbuf[cur][0][0];
    const _Float16* Vc = &VTbuf[cur][0][0];

    // ---- S tiles: Q (global/L2) x K^T (LDS) ----
    v8f s0 = vzero8(), s1 = vzero8();
    const int n0 = nh * 2;
    const _Float16* k0 = Kc + (size_t)(n0 * 16 + ln) * 520;
    const _Float16* k1 = k0 + 16 * 520;
#pragma unroll 4
    for (int kk = 0; kk < DDIM / 32; ++kk) {
      const int kb = kk * 32;
      v16h a  = load_a16(qrow, kb + hb * 8);
      v16h b0 = load_b16(k0 + kb + hb * 8);
      v16h b1 = load_b16(k1 + kb + hb * 8);
      s0 = WMMA_F16(a, b0, s0);
      s1 = WMMA_F16(a, b1, s1);
    }
#pragma unroll
    for (int v = 0; v < 8; ++v) {
      const int r = mi * 16 + v + hb * 8;
      S_t[r][n0 * 16 + ln]       = s0[v];
      S_t[r][(n0 + 1) * 16 + ln] = s1[v];
    }
    __syncthreads();

    // ---- online softmax over the 64-wide tile ----
    {
      const int row = t & 63, seg = t >> 6;
      float mx = S_t[row][seg * 16];
#pragma unroll
      for (int c = 1; c < 16; ++c) mx = fmaxf(mx, S_t[row][seg * 16 + c]);
      red[seg][row] = mx;
    }
    __syncthreads();
    if (t < 64) {
      const float m_old = m_row[t];
      float mx = fmaxf(fmaxf(red[0][t], red[1][t]), fmaxf(red[2][t], red[3][t]));
      const float m_new = fmaxf(m_old, mx);
      m_row[t] = m_new;
      a_row[t] = __expf(m_old - m_new);    // first iter: exp(-1e30) == 0
    }
    __syncthreads();
    {
      const int row = t & 63, seg = t >> 6;
      const float m = m_row[row];
      float sum = 0.0f;
#pragma unroll
      for (int c = 0; c < 16; ++c) {
        const float e = __expf(S_t[row][seg * 16 + c] - m);
        sum += e;
        P_t[row][seg * 16 + c] = (_Float16)e;
      }
      red[seg][row] = sum;
    }
    __syncthreads();
    if (t < 64)
      l_row[t] = l_row[t] * a_row[t] + red[0][t] + red[1][t] + red[2][t] + red[3][t];

    // ---- rescale accumulators, then O += P (LDS) x V (LDS) ----
    float av[8];
#pragma unroll
    for (int v = 0; v < 8; ++v) av[v] = a_row[mi * 16 + v + hb * 8];
#pragma unroll
    for (int ct = 0; ct < 16; ++ct)
#pragma unroll
      for (int v = 0; v < 8; ++v) acc[ct][v] *= av[v];

    const _Float16* prow = &P_t[mi * 16 + ln][0];
    v16h pa0 = load_a16(prow, 0  + hb * 8);
    v16h pa1 = load_a16(prow, 32 + hb * 8);
    const int cbase = nh * 256;
#pragma unroll 4
    for (int ct = 0; ct < 16; ++ct) {
      const int feat = cbase + ct * 16 + ln;
      const _Float16* vp = Vc + (size_t)feat * 72 + hb * 8;
      v16h b0 = load_b16(vp);
      v16h b1 = load_b16(vp + 32);
      acc[ct] = WMMA_F16(pa0, b0, acc[ct]);
      acc[ct] = WMMA_F16(pa1, b1, acc[ct]);
    }
    __syncthreads();                       // tile j fully consumed; buffer reusable
  }

  // ---- epilogue: out = acc / l / sqrt(D)  (softmax-then-scale quirk) ----
  __syncthreads();
  const float scale = 0.044194173824159216f;   // 1/sqrt(512)
  float inv[8];
#pragma unroll
  for (int v = 0; v < 8; ++v) inv[v] = scale / l_row[mi * 16 + v + hb * 8];
  float* outB = out + (size_t)batch * S_LEN * DDIM;
#pragma unroll
  for (int ct = 0; ct < 16; ++ct) {
    const int col = nh * 256 + ct * 16 + ln;
#pragma unroll
    for (int v = 0; v < 8; ++v) {
      const int row = qb + mi * 16 + v + hb * 8;
      outB[(size_t)row * DDIM + col] = acc[ct][v] * inv[v];
    }
  }
}

extern "C" void kernel_launch(void* const* d_in, const int* in_sizes, int n_in,
                              void* d_out, int out_size, void* d_ws, size_t ws_size,
                              hipStream_t stream) {
  const float* x = (const float*)d_in[0];     // (4,4096,512) f32
  const float* W = (const float*)d_in[1];     // (512,512)    f32
  const float* b = (const float*)d_in[2];     // (512,)       f32
  float* out = (float*)d_out;                 // (4,4096,512) f32

  char* ws = (char*)d_ws;
  _Float16* WT   = (_Float16*)ws;                               // 0.5 MiB
  _Float16* h16  = (_Float16*)(ws + (512u * 512u * 2u));        // 16 MiB
  _Float16* hT16 = (_Float16*)(ws + (512u * 512u * 2u) + ((size_t)16384 * 512 * 2));

  wt_convert_kernel<<<1024, 256, 0, stream>>>(W, WT);
  proj_kernel<<<2048, 256, 0, stream>>>(x, WT, b, h16, hT16);
  attn_kernel<<<256, 256, 0, stream>>>(h16, hT16, out);
}
